// PredictiveGRU_46866683134693
// MI455X (gfx1250) — compile-verified, matrix operand order checked by reading, not run on hardware
//
#include <hip/hip_runtime.h>
#include <hip/hip_bf16.h>
#include <math.h>

#define B_ 64
#define L_ 1024
#define D_ 64
#define H_ 512
#define NWG 16        // recurrence workgroups, each owns a 32-wide j-slice of H
#define KT_TOT 18     // K tiles of 32: 2 for x (K=64) + 16 for h (K=512)

typedef __attribute__((ext_vector_type(16))) _Float16 v16h;
typedef __attribute__((ext_vector_type(8)))  _Float16 v8h;
typedef __attribute__((ext_vector_type(8)))  float    v8f;
typedef __attribute__((ext_vector_type(4)))  float    v4f;

// workspace layout (bytes)
#define WS_CNT   0
#define WS_HBUF  256                                  // 2 * 64*512 f16 = 131072 B
#define WS_WSW   (WS_HBUF + 2*B_*H_*2)                // 96*18*32*16 f16 = 1769472 B
#define WS_WPRED (WS_WSW + 96*KT_TOT*32*16*2)         // 4*16*32*16 f16 = 65536 B
#define WS_XSW   (WS_WPRED + 4*16*32*16*2)            // 1024*4*2*32*16 f16 = 8388608 B

__device__ __forceinline__ v8f wmma16(v16h a, v16h b, v8f c) {
  return __builtin_amdgcn_wmma_f32_16x16x32_f16(false, a, false, b, (short)0, c, false, false);
}

// ---------------- init: h ping-pong plane 0 from h0, zero barrier counter ----------------
__global__ void init_kernel(const float* __restrict__ h0, _Float16* __restrict__ hbuf0,
                            unsigned* __restrict__ cnt) {
  int i = blockIdx.x * blockDim.x + threadIdx.x;
  if (i == 0) *cnt = 0u;
  if (i < B_ * H_) hbuf0[i] = (_Float16)h0[i];
}

// ---------------- prep: swizzle [W_ih | W_hh] into per-lane WMMA B fragments ----------------
// layout: [ntile 0..95][kt 0..17][lane 0..31][16 halves]
// element i of lane L = W[ntile*16 + (L&15)][ kt_base + kmap(i, L>>4) ]
__global__ void prep_gates(const float* __restrict__ Wih, const float* __restrict__ Whh,
                           _Float16* __restrict__ wsw) {
  int idx = blockIdx.x * blockDim.x + threadIdx.x;  // (nt*18 + kt)*32 + lane
  if (idx >= 96 * KT_TOT * 32) return;
  int lane = idx & 31;
  int kt   = (idx >> 5) % KT_TOT;
  int nt   = idx / (32 * KT_TOT);
  int row  = nt * 16 + (lane & 15);
  int lh   = lane >> 4;
  _Float16* out = wsw + (size_t)idx * 16;
#pragma unroll
  for (int i = 0; i < 16; ++i) {
    int kl = ((i < 8) ? 0 : 16) + lh * 8 + (i & 7);
    float v = (kt < 2) ? Wih[row * D_ + kt * 32 + kl]
                       : Whh[row * H_ + (kt - 2) * 32 + kl];
    out[i] = (_Float16)v;
  }
}

// ---------------- prep: swizzle W_pred (D x H) into B fragments: [nt 0..3][kt 0..15][lane][16]
__global__ void prep_pred(const float* __restrict__ Wp, _Float16* __restrict__ wp) {
  int idx = blockIdx.x * blockDim.x + threadIdx.x;
  if (idx >= 4 * 16 * 32) return;
  int lane = idx & 31;
  int kt   = (idx >> 5) & 15;
  int nt   = idx / (32 * 16);
  int row  = nt * 16 + (lane & 15);
  int lh   = lane >> 4;
  _Float16* out = wp + (size_t)idx * 16;
#pragma unroll
  for (int i = 0; i < 16; ++i) {
    int kl = ((i < 8) ? 0 : 16) + lh * 8 + (i & 7);
    out[i] = (_Float16)Wp[row * H_ + kt * 32 + kl];
  }
}

// ---------------- prep: swizzle x into f16 WMMA A fragments ----------------
// layout: [t][mt 0..3][kt 0..1][lane][16] -> one 32B v16h load per k-tile in the hot loop
__global__ void prep_x(const float* __restrict__ x, _Float16* __restrict__ xsw) {
  int idx = blockIdx.x * blockDim.x + threadIdx.x;  // ((t*4+mt)*2+kt)*32+lane
  if (idx >= L_ * 4 * 2 * 32) return;
  int lane = idx & 31;
  int kt   = (idx >> 5) & 1;
  int mt   = (idx >> 6) & 3;
  int t    = idx >> 8;
  int b    = mt * 16 + (lane & 15);
  int lh   = lane >> 4;
  const float* xrow = x + ((size_t)b * L_ + t) * D_;
  _Float16* out = xsw + (size_t)idx * 16;
#pragma unroll
  for (int i = 0; i < 16; ++i) {
    int kl = ((i < 8) ? 0 : 16) + lh * 8 + (i & 7);
    out[i] = (_Float16)xrow[kt * 32 + kl];
  }
}

// ---------------- persistent GRU recurrence ----------------
// 16 workgroups x 256 threads (8 waves, 2 per SIMD for latency hiding).
// WG g owns hidden cols [g*32, g*32+32). wave = (M-tile, ng): 16 batches x 16 j-cols.
// Gate tiles r/z/n for the same j live in one wave: GRU update runs on
// register-resident C fragments; own h tile carried in registers across steps;
// full h exchanged via L2 (f16 ping-pong) with one agent-scope barrier per step.
__global__ void __launch_bounds__(256) gru_kernel(
    const _Float16* __restrict__ xsw, const float* __restrict__ h0,
    const float* __restrict__ b_ih, const float* __restrict__ b_hh,
    const _Float16* __restrict__ wsw, _Float16* __restrict__ hbuf,
    unsigned* __restrict__ cnt, float* __restrict__ hseq, float* __restrict__ hN) {
  __shared__ _Float16 Wlds[6 * KT_TOT * 32 * 16];  // 110592 B, resident all 1024 steps

  const int g    = blockIdx.x;
  const int lane = threadIdx.x & 31;
  const int wave = threadIdx.x >> 5;
  const int mt   = wave >> 1;          // M-tile (batch group of 16)
  const int ng   = wave & 1;           // which 16-wide j sub-slice
  const int l16  = lane & 15;
  const int lh   = lane >> 4;

  // one-time copy of this WG's swizzled weight slice
  // local ntile l = gate*2 + ng  <-  global ntile gate*32 + 2g + ng
  {
    uint4* dst = (uint4*)Wlds;
    for (int c = threadIdx.x; c < 6912; c += blockDim.x) {
      int l = c / 1152;                // 18*32*16 halves = 1152 uint4 per ntile
      int off = c % 1152;
      int gate = l >> 1, sub = l & 1;
      const uint4* src = (const uint4*)(wsw + (size_t)((gate * 32 + 2 * g + sub) * KT_TOT * 32) * 16);
      dst[c] = src[off];
    }
  }
  __syncthreads();

  const int j  = g * 32 + ng * 16 + l16;       // this lane's hidden column
  const float br  = b_ih[j] + b_hh[j];
  const float bz  = b_ih[512 + j] + b_hh[512 + j];
  const float bni = b_ih[1024 + j];
  const float bnh = b_hh[1024 + j];

  float hprev[8];
#pragma unroll
  for (int v = 0; v < 8; ++v) {
    int bb = mt * 16 + v + 8 * lh;             // C layout: M = v + 8*(lane>=16)
    hprev[v] = h0[bb * H_ + j];
  }

  const v16h* WB   = (const v16h*)Wlds;
  const v16h* XS   = (const v16h*)xsw;
  const int   brow = mt * 16 + l16;            // A layout: M = lane&15
  const int   lr = (0 * 2 + ng) * KT_TOT;      // local B-fragment row bases
  const int   lz = (1 * 2 + ng) * KT_TOT;
  const int   ln = (2 * 2 + ng) * KT_TOT;

  for (int t = 0; t < L_; ++t) {
    const _Float16* hrd = hbuf + (size_t)(t & 1) * (B_ * H_);
    _Float16*       hwr = hbuf + (size_t)((t + 1) & 1) * (B_ * H_);

    v8f accr, accz, accni, accnh;
#pragma unroll
    for (int v = 0; v < 8; ++v) { accr[v] = br; accz[v] = bz; accni[v] = bni; accnh[v] = bnh; }

    const _Float16* hrow = hrd + (size_t)brow * H_;
    const v16h* xfrag = XS + ((size_t)(t * 4 + mt) * 2) * 32 + lane;
    __builtin_prefetch(xfrag + 8 * 32, 0, 1);  // next timestep's x fragments

    // x contribution (K tiles 0..1, W_ih) — preswizzled f16 A fragments
#pragma unroll
    for (int kt = 0; kt < 2; ++kt) {
      v16h a = xfrag[kt * 32];
      accr  = wmma16(a, WB[(lr + kt) * 32 + lane], accr);
      accz  = wmma16(a, WB[(lz + kt) * 32 + lane], accz);
      accni = wmma16(a, WB[(ln + kt) * 32 + lane], accni);
    }
    // h contribution (K tiles 2..17, W_hh)
#pragma unroll 4
    for (int kt = 2; kt < KT_TOT; ++kt) {
      int k0 = (kt - 2) * 32 + lh * 8;
      v8h c0 = *(const v8h*)(hrow + k0);
      v8h c1 = *(const v8h*)(hrow + k0 + 16);
      v16h a;
#pragma unroll
      for (int i = 0; i < 8; ++i) { a[i] = c0[i]; a[8 + i] = c1[i]; }
      accr  = wmma16(a, WB[(lr + kt) * 32 + lane], accr);
      accz  = wmma16(a, WB[(lz + kt) * 32 + lane], accz);
      accnh = wmma16(a, WB[(ln + kt) * 32 + lane], accnh);
    }

    // gates + state update on register-resident fragments (v_exp + v_rcp forms)
#pragma unroll
    for (int v = 0; v < 8; ++v) {
      float r  = __builtin_amdgcn_rcpf(1.0f + __expf(-accr[v]));
      float z  = __builtin_amdgcn_rcpf(1.0f + __expf(-accz[v]));
      float a2 = accni[v] + r * accnh[v];
      float n  = 1.0f - 2.0f * __builtin_amdgcn_rcpf(__expf(2.0f * a2) + 1.0f);
      float hn = (1.0f - z) * n + z * hprev[v];
      hprev[v] = hn;
      int bb = mt * 16 + v + 8 * lh;
      hwr[(size_t)bb * H_ + j] = (_Float16)hn;                // for other WGs next step
      hseq[((size_t)bb * L_ + t) * H_ + j] = hn;              // raw; LN pass normalizes in place
      if (t == L_ - 1) hN[(size_t)bb * H_ + j] = hn;
    }

    // device-scope step barrier across the 16 persistent workgroups
    __threadfence();
    __syncthreads();
    if (threadIdx.x == 0)
      __hip_atomic_fetch_add(cnt, 1u, __ATOMIC_RELEASE, __HIP_MEMORY_SCOPE_AGENT);
    unsigned target = (unsigned)NWG * (unsigned)(t + 1);
    while (__hip_atomic_load(cnt, __ATOMIC_RELAXED, __HIP_MEMORY_SCOPE_AGENT) < target)
      __builtin_amdgcn_s_sleep(1);
    __builtin_amdgcn_fence(__ATOMIC_ACQUIRE, "agent");
  }
}

// ---------------- LayerNorm in place over h_seq rows (wave per row) ----------------
__global__ void __launch_bounds__(256) ln_kernel(float* __restrict__ hseq,
                                                 const float* __restrict__ gamma,
                                                 const float* __restrict__ beta) {
  int wave = threadIdx.x >> 5, lane = threadIdx.x & 31;
  size_t row = (size_t)blockIdx.x * 8 + wave;
  float* p = hseq + row * H_;
  v4f vals[4];
  float s = 0.f, sq = 0.f;
#pragma unroll
  for (int c = 0; c < 4; ++c) {
    vals[c] = *(const v4f*)(p + c * 128 + lane * 4);
#pragma unroll
    for (int i = 0; i < 4; ++i) { s += vals[c][i]; sq += vals[c][i] * vals[c][i]; }
  }
#pragma unroll
  for (int m = 16; m >= 1; m >>= 1) { s += __shfl_xor(s, m, 32); sq += __shfl_xor(sq, m, 32); }
  float mu  = s * (1.0f / 512.0f);
  float var = sq * (1.0f / 512.0f) - mu * mu;
  float inv = rsqrtf(var + 1e-5f);
#pragma unroll
  for (int c = 0; c < 4; ++c) {
    int d0 = c * 128 + lane * 4;
    v4f gv = *(const v4f*)(gamma + d0);
    v4f bv = *(const v4f*)(beta + d0);
    v4f o;
#pragma unroll
    for (int i = 0; i < 4; ++i) o[i] = (vals[c][i] - mu) * inv * gv[i] + bv[i];
    *(v4f*)(p + d0) = o;
  }
}

// ---------------- prediction GEMM: x_hat = h_seq @ W_pred^T + b_pred ----------------
// M = B*L = 65536, N = 64 (4 n-tiles), K = 512 (16 k-tiles); wave per M-tile.
__global__ void __launch_bounds__(256) pred_kernel(const float* __restrict__ hseq,
                                                   const _Float16* __restrict__ wp,
                                                   const float* __restrict__ bpred,
                                                   float* __restrict__ xhat) {
  int lane = threadIdx.x & 31, wave = threadIdx.x >> 5;
  int l16 = lane & 15, lh = lane >> 4;
  size_t mt = (size_t)blockIdx.x * 8 + wave;  // < 4096
  const float* arow = hseq + (mt * 16 + l16) * (size_t)H_;
  const v16h* WP = (const v16h*)wp;
  v8f acc[4];
#pragma unroll
  for (int nt = 0; nt < 4; ++nt) {
    float bv = bpred[nt * 16 + l16];
#pragma unroll
    for (int v = 0; v < 8; ++v) acc[nt][v] = bv;
  }
#pragma unroll 4
  for (int kt = 0; kt < 16; ++kt) {
    int k0 = kt * 32 + lh * 8;
    v4f p0 = *(const v4f*)(arow + k0);
    v4f p1 = *(const v4f*)(arow + k0 + 4);
    v4f p2 = *(const v4f*)(arow + k0 + 16);
    v4f p3 = *(const v4f*)(arow + k0 + 20);
    v16h a;
#pragma unroll
    for (int i = 0; i < 4; ++i) {
      a[i] = (_Float16)p0[i]; a[4 + i] = (_Float16)p1[i];
      a[8 + i] = (_Float16)p2[i]; a[12 + i] = (_Float16)p3[i];
    }
#pragma unroll
    for (int nt = 0; nt < 4; ++nt) acc[nt] = wmma16(a, WP[(nt * 16 + kt) * 32 + lane], acc[nt]);
  }
#pragma unroll
  for (int nt = 0; nt < 4; ++nt)
#pragma unroll
    for (int v = 0; v < 8; ++v)
      xhat[(mt * 16 + v + 8 * lh) * (size_t)D_ + nt * 16 + l16] = acc[nt][v];
}

extern "C" void kernel_launch(void* const* d_in, const int* in_sizes, int n_in,
                              void* d_out, int out_size, void* d_ws, size_t ws_size,
                              hipStream_t stream) {
  const float* x     = (const float*)d_in[0];
  const float* h0    = (const float*)d_in[1];
  const float* Wih   = (const float*)d_in[2];
  const float* Whh   = (const float*)d_in[3];
  const float* bih   = (const float*)d_in[4];
  const float* bhh   = (const float*)d_in[5];
  const float* gamma = (const float*)d_in[6];
  const float* beta  = (const float*)d_in[7];
  const float* Wp    = (const float*)d_in[8];
  const float* bp    = (const float*)d_in[9];

  float* out  = (float*)d_out;
  float* hseq = out;                              // (B, L, H)
  float* hN   = out + (size_t)B_ * L_ * H_;       // (1, B, H)
  float* xhat = hN + (size_t)B_ * H_;             // (B, L, D)

  char* ws = (char*)d_ws;
  unsigned* cnt  = (unsigned*)(ws + WS_CNT);
  _Float16* hbuf = (_Float16*)(ws + WS_HBUF);
  _Float16* wsw  = (_Float16*)(ws + WS_WSW);
  _Float16* wp   = (_Float16*)(ws + WS_WPRED);
  _Float16* xsw  = (_Float16*)(ws + WS_XSW);

  init_kernel<<<(B_ * H_ + 255) / 256, 256, 0, stream>>>(h0, hbuf, cnt);
  prep_gates<<<(96 * KT_TOT * 32 + 255) / 256, 256, 0, stream>>>(Wih, Whh, wsw);
  prep_pred<<<(4 * 16 * 32 + 255) / 256, 256, 0, stream>>>(Wp, wp);
  prep_x<<<(L_ * 4 * 2 * 32 + 255) / 256, 256, 0, stream>>>(x, xsw);
  gru_kernel<<<NWG, 256, 0, stream>>>(xsw, h0, bih, bhh, wsw, hbuf, cnt, hseq, hN);
  ln_kernel<<<(B_ * L_) / 8, 256, 0, stream>>>(hseq, gamma, beta);
  pred_kernel<<<(B_ * L_ / 16) / 8, 256, 0, stream>>>(hseq, wp, bp, xhat);
}
